// TriangleAttentionEndingNode_30270929502440
// MI455X (gfx1250) — compile-verified, matrix operand order checked by reading, not run on hardware
//
#include <hip/hip_runtime.h>

// ---------------------------------------------------------------------------
// TriangleAttentionEndingNode for MI455X (gfx1250, wave32, WMMA bf16)
//   M = N = 384, C_IN = 128, H = 4, C_ATT = 32  (head dim == WMMA K: perfect)
// ---------------------------------------------------------------------------

#define MRES   384
#define CIN    128
#define NHEADS 4
#define CATT   32
#define HC     128
#define LN_EPS 1e-5f
#define QSCALE 0.17677669529663687f   // 1/sqrt(32)

typedef __attribute__((ext_vector_type(4)))  float  f32x4;
typedef __attribute__((ext_vector_type(8)))  float  f32x8;
typedef __attribute__((ext_vector_type(4)))  __bf16 bf16x4;
typedef __attribute__((ext_vector_type(8)))  __bf16 bf16x8;
typedef __attribute__((ext_vector_type(16))) __bf16 bf16x16;

// D = A(16x32 bf16) * B(32x16 bf16) + C(16x16 f32), one wave.
__device__ __forceinline__ f32x8 wmma_bf16(bf16x16 a, bf16x16 b, f32x8 c) {
  return __builtin_amdgcn_wmma_f32_16x16x32_bf16(false, a, false, b,
                                                 (short)0, c, false, false);
}

__device__ __forceinline__ float fast_rcp(float x) {
  return __builtin_amdgcn_rcpf(x);    // v_rcp_f32, avoids div_scale sequence
}

// Load a 16x32 bf16 WMMA operand (A, or B supplied as B^T rows) from a
// row-major [16][ld] source. Per ISA 7.12.2 (16-bit A 16x32):
//   lane L: row = L&15; K elements {k0..k0+7} and {k0+16..k0+23}, k0 = (L<16?0:8)
__device__ __forceinline__ bf16x16 load_op_16x32(const __bf16* base, int ld, int lane) {
  int row = lane & 15;
  int k0  = (lane < 16) ? 0 : 8;
  const __bf16* p = base + (size_t)row * ld + k0;
  bf16x8 lo = *(const bf16x8*)(p);
  bf16x8 hi = *(const bf16x8*)(p + 16);
  bf16x16 r;
#pragma unroll
  for (int t = 0; t < 8; ++t) { r[t] = lo[t]; r[t + 8] = hi[t]; }
  return r;
}

// ---------------------------------------------------------------------------
// Kernel W: repack weights to bf16 [out][in] (contiguous per-lane B operands).
//   wqkvgT[512][128]  (sel-major: q,k,v,g) ;  woT[128][128]
// ---------------------------------------------------------------------------
__global__ void kW_pack(const float* __restrict__ Wq, const float* __restrict__ Wk,
                        const float* __restrict__ Wv, const float* __restrict__ Wg,
                        const float* __restrict__ Wo,
                        __bf16* __restrict__ wqkvgT, __bf16* __restrict__ woT) {
  int idx = blockIdx.x * 256 + threadIdx.x;     // 0 .. 512*128-1
  int o = idx >> 7, i = idx & 127;
  int sel = o >> 7, oc = o & 127;
  const float* W = (sel == 0) ? Wq : (sel == 1) ? Wk : (sel == 2) ? Wv : Wg;
  wqkvgT[idx] = (__bf16)W[i * HC + oc];
  if (idx < HC * CIN) {                          // woT[o][i] = Wo[i][o]
    woT[idx] = (__bf16)Wo[i * CIN + o];
  }
}

// ---------------------------------------------------------------------------
// Kernel A: LayerNorm of x = swapaxes(x2d) into bf16 xln[m][j][128],
//           plus pair bias b[h][i][j] = ln(x)[i,j,:] . Wb[:,h]  (fp32).
// One wave per row; 32 lanes x float4.
// ---------------------------------------------------------------------------
__global__ __launch_bounds__(256)
void kA_ln(const float* __restrict__ x2d, const float* __restrict__ ln_s,
           const float* __restrict__ ln_b, const float* __restrict__ Wb,
           __bf16* __restrict__ xln, float* __restrict__ bbuf) {
  int wave = threadIdx.x >> 5, lane = threadIdx.x & 31;
  int row  = blockIdx.x * 8 + wave;              // row = m*384 + j
  int m = row / MRES, j = row - m * MRES;

  // x[m, j, c] = x2d[0, j, m, c]  (swapaxes)
  const float* src = x2d + ((size_t)j * MRES + m) * CIN;
  f32x4 x = *(const f32x4*)(src + lane * 4);

  float s1 = x[0] + x[1] + x[2] + x[3];
  float s2 = x[0]*x[0] + x[1]*x[1] + x[2]*x[2] + x[3]*x[3];
#pragma unroll
  for (int msk = 1; msk < 32; msk <<= 1) {
    s1 += __shfl_xor(s1, msk, 32);
    s2 += __shfl_xor(s2, msk, 32);
  }
  float mu  = s1 * (1.0f / CIN);
  float var = s2 * (1.0f / CIN) - mu * mu;
  float rs  = rsqrtf(var + LN_EPS);

  f32x4 sc = *(const f32x4*)(ln_s + lane * 4);
  f32x4 bi = *(const f32x4*)(ln_b + lane * 4);
  f32x4 y;
  bf16x4 yb;
#pragma unroll
  for (int t = 0; t < 4; ++t) {
    y[t]  = (x[t] - mu) * rs * sc[t] + bi[t];
    yb[t] = (__bf16)y[t];
  }
  *(bf16x4*)(xln + (size_t)row * CIN + lane * 4) = yb;   // 8B per lane, coalesced

  // bias: acc[h] = sum_c y[c] * Wb[c][h]
  f32x4 acc = {0.f, 0.f, 0.f, 0.f};
#pragma unroll
  for (int t = 0; t < 4; ++t) {
    f32x4 w = *(const f32x4*)(Wb + (size_t)(lane * 4 + t) * NHEADS);
    acc[0] += y[t] * w[0]; acc[1] += y[t] * w[1];
    acc[2] += y[t] * w[2]; acc[3] += y[t] * w[3];
  }
#pragma unroll
  for (int msk = 1; msk < 32; msk <<= 1) {
#pragma unroll
    for (int t = 0; t < 4; ++t) acc[t] += __shfl_xor(acc[t], msk, 32);
  }
  if (lane == 0) {
#pragma unroll
    for (int h = 0; h < NHEADS; ++h)
      bbuf[((size_t)h * MRES + m) * MRES + j] = acc[h];
  }
}

// ---------------------------------------------------------------------------
// Kernel B: per (m, h): head-sliced q,k,v,g projections (WMMA -> LDS), then
// flash attention with bias, gating, bf16 output o[m][i][h*32+c].
// 256 threads = 8 waves. Each wave owns one (sel, ct) projection stream:
// weight B-tiles are register-resident, xln A-tiles are double-buffered.
// ---------------------------------------------------------------------------
__global__ __launch_bounds__(256)
void kB_attn(const __bf16* __restrict__ xln, const __bf16* __restrict__ wqkvgT,
             const float* __restrict__ bbuf, const float* __restrict__ bg,
             __bf16* __restrict__ obuf) {
  __shared__ __bf16 sQ[MRES * CATT];     // [i][c]   24 KB (pre-scaled)
  __shared__ __bf16 sK[MRES * CATT];     // [j][c]   24 KB
  __shared__ __bf16 sVT[CATT * MRES];    // [c][j]   24 KB  (transposed V)
  __shared__ __bf16 sG[MRES * CATT];     // [i][c]   24 KB  (sigmoid applied)
  __shared__ __bf16 sP[8 * 16 * CATT];   // per-wave P scratch, 8 KB

  int m = blockIdx.x >> 2;
  int h = blockIdx.x & 3;
  int wave = threadIdx.x >> 5, lane = threadIdx.x & 31;
  int col = lane & 15, rbase = (lane < 16) ? 0 : 8;

  // ---- Phase 1: q/k/v/g head slices: [384,128] @ wT[sel*128+h*32+ct*16 ..]
  // wave -> fixed (sel, ct); iterate the 24 j-tiles with pipelined A loads.
  {
    int sel = wave & 3;
    int ct  = (wave >> 2) & 1;
    const __bf16* wbase = wqkvgT + (size_t)(sel * HC + h * CATT + ct * 16) * CIN;
    const __bf16* arow  = xln + (size_t)m * MRES * CIN;
    float bgv = bg[h * CATT + ct * 16 + col];   // only used when sel==3

    bf16x16 wb[4];
#pragma unroll
    for (int kk = 0; kk < 4; ++kk)
      wb[kk] = load_op_16x32(wbase + kk * 32, CIN, lane);

    bf16x16 a_nxt[4];
#pragma unroll
    for (int kk = 0; kk < 4; ++kk)
      a_nxt[kk] = load_op_16x32(arow + kk * 32, CIN, lane);

    for (int jt = 0; jt < 24; ++jt) {
      bf16x16 a[4];
#pragma unroll
      for (int kk = 0; kk < 4; ++kk) a[kk] = a_nxt[kk];
      int jn = (jt + 1 < 24) ? (jt + 1) : 23;   // clamped prefetch (uniform, no divergence)
      const __bf16* nb = arow + (size_t)jn * 16 * CIN;
#pragma unroll
      for (int kk = 0; kk < 4; ++kk)
        a_nxt[kk] = load_op_16x32(nb + kk * 32, CIN, lane);

      f32x8 c = {};
#pragma unroll
      for (int kk = 0; kk < 4; ++kk) c = wmma_bf16(a[kk], wb[kk], c);

      int r0 = jt * 16 + rbase;
      int c0 = ct * 16 + col;
      if (sel == 0) {
#pragma unroll
        for (int r = 0; r < 8; ++r)
          sQ[(r0 + r) * CATT + c0] = (__bf16)(c[r] * QSCALE);
      } else if (sel == 1) {
#pragma unroll
        for (int r = 0; r < 8; ++r)
          sK[(r0 + r) * CATT + c0] = (__bf16)c[r];
      } else if (sel == 2) {
#pragma unroll
        for (int r = 0; r < 8; ++r)
          sVT[(size_t)c0 * MRES + r0 + r] = (__bf16)c[r];
      } else {
#pragma unroll
        for (int r = 0; r < 8; ++r) {
          float e = __expf(-(c[r] + bgv));
          sG[(r0 + r) * CATT + c0] = (__bf16)fast_rcp(1.0f + e);   // sigmoid
        }
      }
    }
  }
  __syncthreads();

  // ---- Phase 2: flash attention, 24 i-tiles over 8 waves
  const float* brow = bbuf + (size_t)h * MRES * MRES;
  __bf16* ps = sP + wave * 16 * CATT;
  for (int it = wave; it < 24; it += 8) {
    int i0 = it * 16;
    bf16x16 aQ = load_op_16x32(sQ + (size_t)i0 * CATT, CATT, lane);  // K=c=32: single A tile
    f32x8 acc0 = {}, acc1 = {};
    float mrow[8], lrow[8];
#pragma unroll
    for (int r = 0; r < 8; ++r) { mrow[r] = -1e30f; lrow[r] = 0.0f; }

    for (int jc = 0; jc < 12; ++jc) {
      int j0 = jc * 32;
      bf16x16 bK0 = load_op_16x32(sK + (size_t)j0 * CATT, CATT, lane);
      bf16x16 bK1 = load_op_16x32(sK + (size_t)(j0 + 16) * CATT, CATT, lane);
      f32x8 z = {};
      f32x8 s0 = wmma_bf16(aQ, bK0, z);
      f32x8 s1 = wmma_bf16(aQ, bK1, z);

      // + pair bias, then online softmax update (rows live in one 16-lane half)
#pragma unroll
      for (int r = 0; r < 8; ++r) {
        const float* bp = brow + (size_t)(i0 + rbase + r) * MRES + j0;
        s0[r] += bp[col];
        s1[r] += bp[16 + col];
        float v = fmaxf(s0[r], s1[r]);
        v = fmaxf(v, __shfl_xor(v, 1, 32));
        v = fmaxf(v, __shfl_xor(v, 2, 32));
        v = fmaxf(v, __shfl_xor(v, 4, 32));
        v = fmaxf(v, __shfl_xor(v, 8, 32));
        float mnew  = fmaxf(mrow[r], v);
        float corr  = __expf(mrow[r] - mnew);
        mrow[r] = mnew;
        float e0 = __expf(s0[r] - mnew);
        float e1 = __expf(s1[r] - mnew);
        float rs = e0 + e1;
        rs += __shfl_xor(rs, 1, 32);
        rs += __shfl_xor(rs, 2, 32);
        rs += __shfl_xor(rs, 4, 32);
        rs += __shfl_xor(rs, 8, 32);
        lrow[r] = lrow[r] * corr + rs;
        acc0[r] = acc0[r] * corr;
        acc1[r] = acc1[r] * corr;
        s0[r] = e0;
        s1[r] = e1;
      }
      // P (C layout) -> LDS -> A layout
#pragma unroll
      for (int r = 0; r < 8; ++r) {
        ps[(rbase + r) * CATT + col]      = (__bf16)s0[r];
        ps[(rbase + r) * CATT + 16 + col] = (__bf16)s1[r];
      }
      bf16x16 aP  = load_op_16x32(ps, CATT, lane);
      bf16x16 bV0 = load_op_16x32(sVT + j0, MRES, lane);                      // c-tile 0
      bf16x16 bV1 = load_op_16x32(sVT + (size_t)16 * MRES + j0, MRES, lane);  // c-tile 1
      acc0 = wmma_bf16(aP, bV0, acc0);
      acc1 = wmma_bf16(aP, bV1, acc1);
    }

    // finalize: /l, gate with sigmoid(g), store bf16 o[m][i][h*32+c]
#pragma unroll
    for (int r = 0; r < 8; ++r) {
      int irow = i0 + rbase + r;
      float inv = fast_rcp(lrow[r]);
      float g0 = (float)sG[irow * CATT + col];
      float g1 = (float)sG[irow * CATT + 16 + col];
      size_t ob = ((size_t)m * MRES + irow) * HC + h * CATT;
      obuf[ob + col]      = (__bf16)(acc0[r] * inv * g0);
      obuf[ob + 16 + col] = (__bf16)(acc1[r] * inv * g1);
    }
  }
}

// ---------------------------------------------------------------------------
// Kernel C: out = o @ Wo + bo, written with axes swapped back (fp32).
// 8 waves/block, each wave owns a 16-row tile (row = m*384 + i).
// A tiles register-resident; woT B-tiles double-buffered across n-tiles.
// ---------------------------------------------------------------------------
__global__ __launch_bounds__(256)
void kC_proj(const __bf16* __restrict__ obuf, const __bf16* __restrict__ woT,
             const float* __restrict__ bo, float* __restrict__ out) {
  int wave = threadIdx.x >> 5, lane = threadIdx.x & 31;
  int rt = blockIdx.x * 8 + wave;
  int row0 = rt * 16;
  int col = lane & 15, rbase = (lane < 16) ? 0 : 8;

  bf16x16 A[4];
#pragma unroll
  for (int kk = 0; kk < 4; ++kk)
    A[kk] = load_op_16x32(obuf + (size_t)row0 * HC + kk * 32, HC, lane);

  bf16x16 b_nxt[4];
#pragma unroll
  for (int kk = 0; kk < 4; ++kk)
    b_nxt[kk] = load_op_16x32(woT + kk * 32, CIN, lane);

#pragma unroll
  for (int nt = 0; nt < 8; ++nt) {
    bf16x16 b[4];
#pragma unroll
    for (int kk = 0; kk < 4; ++kk) b[kk] = b_nxt[kk];
    int nn = (nt + 1 < 8) ? (nt + 1) : 7;
#pragma unroll
    for (int kk = 0; kk < 4; ++kk)
      b_nxt[kk] = load_op_16x32(woT + (size_t)(nn * 16) * CIN + kk * 32, CIN, lane);

    f32x8 c = {};
#pragma unroll
    for (int kk = 0; kk < 4; ++kk) c = wmma_bf16(A[kk], b[kk], c);

    float bias = bo[nt * 16 + col];
#pragma unroll
    for (int r = 0; r < 8; ++r) {
      int row = row0 + rbase + r;           // row = m*384 + i
      int m = row / MRES, i = row - m * MRES;
      out[((size_t)i * MRES + m) * CIN + nt * 16 + col] = c[r] + bias;  // swapaxes back
    }
  }
}

// ---------------------------------------------------------------------------
extern "C" void kernel_launch(void* const* d_in, const int* in_sizes, int n_in,
                              void* d_out, int out_size, void* d_ws, size_t ws_size,
                              hipStream_t stream) {
  (void)in_sizes; (void)n_in; (void)out_size;
  const float* x2d  = (const float*)d_in[0];
  const float* ln_s = (const float*)d_in[1];
  const float* ln_b = (const float*)d_in[2];
  const float* Wq   = (const float*)d_in[3];
  const float* Wk   = (const float*)d_in[4];
  const float* Wv   = (const float*)d_in[5];
  const float* Wb   = (const float*)d_in[6];
  const float* Wg   = (const float*)d_in[7];
  const float* bg   = (const float*)d_in[8];
  const float* Wo   = (const float*)d_in[9];
  const float* bo   = (const float*)d_in[10];
  float* out = (float*)d_out;

  // workspace layout (bytes, 256-aligned)
  char* ws = (char*)d_ws;
  size_t off = 0;
  auto take = [&](size_t bytes) { char* p = ws + off; off = (off + bytes + 255) & ~(size_t)255; return p; };
  __bf16* wqkvgT = (__bf16*)take((size_t)512 * CIN * 2);            // 128 KB
  __bf16* woT    = (__bf16*)take((size_t)HC * CIN * 2);             //  32 KB
  __bf16* xln    = (__bf16*)take((size_t)MRES * MRES * CIN * 2);    // ~37.7 MB
  float*  bbuf   = (float*)take((size_t)NHEADS * MRES * MRES * 4);  // ~2.4 MB
  __bf16* obuf   = (__bf16*)take((size_t)MRES * MRES * HC * 2);     // ~37.7 MB
  (void)ws_size;  // total ~78 MB; intermediates are L2-resident on MI455X (192 MB L2)

  kW_pack<<<(512 * CIN) / 256, 256, 0, stream>>>(Wq, Wk, Wv, Wg, Wo, wqkvgT, woT);
  kA_ln<<<(MRES * MRES) / 8, 256, 0, stream>>>(x2d, ln_s, ln_b, Wb, xln, bbuf);
  kB_attn<<<MRES * NHEADS, 256, 0, stream>>>(xln, wqkvgT, bbuf, bg, obuf);
  kC_proj<<<(MRES * MRES / 16) / 8, 256, 0, stream>>>(obuf, woT, bo, out);
}
